// AttentionBlock_7687991459972
// MI455X (gfx1250) — compile-verified, matrix operand order checked by reading, not run on hardware
//
#include <hip/hip_runtime.h>

// ---------------- problem constants ----------------
#define BB 2
#define SS 2048
#define DD 1024
#define HH 16
#define HDD 64
#define FFD 4096
#define MM (BB * SS)   // 4096 rows

typedef __attribute__((ext_vector_type(16))) __bf16 v16bf;
typedef __attribute__((ext_vector_type(8)))  float  v8f;

union Frag16 {           // one WMMA 16-bit A/B operand: 16 bf16 = 32B per lane
    v16bf v;
    uint4 q[2];
};

// ---------------- CDNA5 async global->LDS helpers ----------------
typedef __attribute__((ext_vector_type(4))) int i4_t;
typedef __attribute__((address_space(1))) i4_t as1_i4;   // global int4
typedef __attribute__((address_space(3))) i4_t as3_i4;   // LDS int4

#if __has_builtin(__builtin_amdgcn_global_load_async_to_lds_b128)
#define HAS_ASYNC 1
#else
#define HAS_ASYNC 0
#endif

__device__ __forceinline__ void async_b128(const void* g, void* l) {
#if HAS_ASYNC
    __builtin_amdgcn_global_load_async_to_lds_b128((as1_i4*)g, (as3_i4*)l, 0, 0);
#else
    *(uint4*)l = *(const uint4*)g;   // synchronous fallback
#endif
}

__device__ __forceinline__ void wait_async0() {
#if __has_builtin(__builtin_amdgcn_s_wait_asynccnt)
    __builtin_amdgcn_s_wait_asynccnt(0);
#else
    asm volatile("s_wait_asynccnt 0" ::: "memory");
#endif
}

// =====================================================================
// Weight convert + transpose: w[K,N] fp32 -> wt[N,K] bf16
// =====================================================================
__global__ __launch_bounds__(256) void wcvt_kernel(const float* __restrict__ w,
                                                   __bf16* __restrict__ wt,
                                                   int K, int N) {
    __shared__ float tile[32][33];
    int nb = blockIdx.x * 32;
    int kb = blockIdx.y * 32;
    int tx = threadIdx.x & 31;
    int ty = threadIdx.x >> 5;          // 0..7
    for (int i = ty; i < 32; i += 8)
        tile[i][tx] = w[(size_t)(kb + i) * N + nb + tx];
    __syncthreads();
    for (int i = ty; i < 32; i += 8)
        wt[(size_t)(nb + i) * K + kb + tx] = (__bf16)tile[tx][i];
}

// =====================================================================
// LayerNorm (fp32 in -> bf16 out), one block per row
// =====================================================================
__global__ __launch_bounds__(256) void ln_kernel(const float* __restrict__ x,
                                                 const float* __restrict__ scale,
                                                 const float* __restrict__ bias,
                                                 __bf16* __restrict__ out, int D) {
    int row = blockIdx.x;
    const float* xr = x + (size_t)row * D;
    __shared__ float red[256];
    float s = 0.f;
    for (int i = threadIdx.x; i < D; i += 256) s += xr[i];
    red[threadIdx.x] = s;
    __syncthreads();
    for (int st = 128; st > 0; st >>= 1) {
        if (threadIdx.x < st) red[threadIdx.x] += red[threadIdx.x + st];
        __syncthreads();
    }
    float mu = red[0] / (float)D;
    __syncthreads();
    float v = 0.f;
    for (int i = threadIdx.x; i < D; i += 256) {
        float d = xr[i] - mu;
        v += d * d;
    }
    red[threadIdx.x] = v;
    __syncthreads();
    for (int st = 128; st > 0; st >>= 1) {
        if (threadIdx.x < st) red[threadIdx.x] += red[threadIdx.x + st];
        __syncthreads();
    }
    float rs = rsqrtf(red[0] / (float)D + 1e-5f);
    for (int i = threadIdx.x; i < D; i += 256)
        out[(size_t)row * D + i] = (__bf16)((xr[i] - mu) * rs * scale[i] + bias[i]);
}

// =====================================================================
// bf16 WMMA GEMM: C[M,N] = A[M,K] @ Bt[N,K]^T (+bias +GELU +residual)
// 256 threads = 8 waves; block tile 128x128; wave tile 32(M)x64(N)
// BK=64, double-buffered LDS filled by async global->LDS DMA.
// =====================================================================
template <bool GELU_, bool OUTBF>
__global__ __launch_bounds__(256) void gemm_kernel(const __bf16* __restrict__ A,
                                                   const __bf16* __restrict__ Bt,
                                                   const float* __restrict__ bias,
                                                   const float* __restrict__ resid,
                                                   void* __restrict__ Cout,
                                                   int M, int N, int K) {
    __shared__ __align__(16) __bf16 As[2][128 * 64];   // 2 x 16KB
    __shared__ __align__(16) __bf16 Bs[2][128 * 64];   // 2 x 16KB

    int tid  = threadIdx.x;
    int lane = tid & 31;
    int wave = tid >> 5;
    int half = lane >> 4;
    int l16  = lane & 15;
    int wm   = wave >> 1;   // 0..3 -> M strip of 32
    int wn   = wave & 1;    // 0..1 -> N strip of 64
    int bm   = blockIdx.y * 128;
    int bn   = blockIdx.x * 128;

    auto issue_tile = [&](int k0, int buf) {
        for (int i = 0; i < 4; ++i) {
            int linear = (i * 256 + tid) * 8;           // 0..8191 elements
            int row = linear >> 6, col = linear & 63;
            async_b128(&A[(size_t)(bm + row) * K + k0 + col], &As[buf][linear]);
            async_b128(&Bt[(size_t)(bn + row) * K + k0 + col], &Bs[buf][linear]);
        }
    };

    v8f acc[2][4] = {};
    int T = K >> 6;                 // K divisible by 64 for all our GEMMs
    issue_tile(0, 0);
    int buf = 0;

    for (int t = 0; t < T; ++t) {
        wait_async0();              // tile t resident in LDS[buf]
        __syncthreads();            // visible to all waves; LDS[buf^1] reads done
        if (t + 1 < T) issue_tile((t + 1) << 6, buf ^ 1);  // DMA overlaps compute

        for (int kk = 0; kk < 2; ++kk) {
            Frag16 af[2], bfr[4];
            for (int mt = 0; mt < 2; ++mt) {
                int row = wm * 32 + mt * 16 + l16;
                af[mt].q[0] = *(uint4*)&As[buf][row * 64 + kk * 32 + half * 8];
                af[mt].q[1] = *(uint4*)&As[buf][row * 64 + kk * 32 + 16 + half * 8];
            }
            for (int nt = 0; nt < 4; ++nt) {
                int row = wn * 64 + nt * 16 + l16;
                bfr[nt].q[0] = *(uint4*)&Bs[buf][row * 64 + kk * 32 + half * 8];
                bfr[nt].q[1] = *(uint4*)&Bs[buf][row * 64 + kk * 32 + 16 + half * 8];
            }
            for (int mt = 0; mt < 2; ++mt)
                for (int nt = 0; nt < 4; ++nt)
                    acc[mt][nt] = __builtin_amdgcn_wmma_f32_16x16x32_bf16(
                        false, af[mt].v, false, bfr[nt].v, (short)0, acc[mt][nt],
                        false, false);
        }
        buf ^= 1;
    }

    // epilogue
    for (int mt = 0; mt < 2; ++mt) {
        int mbase = bm + wm * 32 + mt * 16 + half * 8;
        for (int nt = 0; nt < 4; ++nt) {
            int n = bn + wn * 64 + nt * 16 + l16;
            float bia = bias ? bias[n] : 0.f;
            for (int r = 0; r < 8; ++r) {
                int m = mbase + r;
                float v = acc[mt][nt][r] + bia;
                if (GELU_) v = 0.5f * v * (1.f + erff(v * 0.70710678f));
                if (resid) v += resid[(size_t)m * N + n];
                if (OUTBF)
                    ((__bf16*)Cout)[(size_t)m * N + n] = (__bf16)v;
                else
                    ((float*)Cout)[(size_t)m * N + n] = v;
            }
        }
    }
}

// =====================================================================
// RoPE + layout reorder:
//   qf/kf/vf fp32 [B*S, H*HD]  ->  qbf/kbf bf16 [B,H,S,HD], vtb bf16 [B,H,HD,S]
//   q additionally scaled by 1/sqrt(HD). ROT == HD == 64 here.
// =====================================================================
__global__ __launch_bounds__(256) void rope_kernel(const float* __restrict__ qf,
                                                   const float* __restrict__ kf,
                                                   const float* __restrict__ vf,
                                                   const int* __restrict__ pos_ids,
                                                   __bf16* __restrict__ qbf,
                                                   __bf16* __restrict__ kbf,
                                                   __bf16* __restrict__ vtb) {
    int row = blockIdx.x;                 // b*S + s
    int b = row / SS, s = row % SS;
    float p = (float)pos_ids[row];
    for (int pp = threadIdx.x; pp < (HH * HDD) / 2; pp += 256) {
        int h = pp / (HDD / 2);
        int j = pp % (HDD / 2);
        int d0 = 2 * j, d1 = 2 * j + 1;
        size_t base = (size_t)row * DD + (size_t)h * HDD;
        float inv = __powf(10000.f, -(float)(2 * j) / 64.f);
        float sn, cs;
        __sincosf(p * inv, &sn, &cs);
        float q0 = qf[base + d0], q1 = qf[base + d1];
        float k0 = kf[base + d0], k1 = kf[base + d1];
        size_t obase = (((size_t)b * HH + h) * SS + s) * HDD;
        qbf[obase + d0] = (__bf16)((q0 * cs - q1 * sn) * 0.125f);
        qbf[obase + d1] = (__bf16)((q1 * cs + q0 * sn) * 0.125f);
        kbf[obase + d0] = (__bf16)(k0 * cs - k1 * sn);
        kbf[obase + d1] = (__bf16)(k1 * cs + k0 * sn);
        size_t vb = ((size_t)b * HH + h) * (size_t)HDD * SS;
        vtb[vb + (size_t)d0 * SS + s] = (__bf16)vf[base + d0];
        vtb[vb + (size_t)d1 * SS + s] = (__bf16)vf[base + d1];
    }
}

// =====================================================================
// Flash attention (causal): grid (S/64, B*H), 128 threads = 4 waves.
// Double-buffered async K/V tiles; P aliases the fp32 score buffer.
// =====================================================================
__global__ __launch_bounds__(128) void attn_kernel(const __bf16* __restrict__ qbf,
                                                   const __bf16* __restrict__ kbf,
                                                   const __bf16* __restrict__ vtb,
                                                   const int* __restrict__ amask,
                                                   __bf16* __restrict__ obf) {
    int qt = blockIdx.x;                  // query tile (64 rows)
    int bh = blockIdx.y;
    int b = bh / HH, h = bh % HH;
    const __bf16* Qp = qbf + (((size_t)bh * SS) + (size_t)qt * 64) * HDD;
    const __bf16* Kp = kbf + (size_t)bh * SS * HDD;
    const __bf16* Vp = vtb + (size_t)bh * HDD * SS;

    __shared__ __align__(16) __bf16 Qs[64 * 64];       // 8KB
    __shared__ __align__(16) __bf16 Ks[2][64 * 64];    // 16KB
    __shared__ __align__(16) __bf16 Vs[2][64 * 64];    // 16KB (rows = hd)
    __shared__ __align__(16) float  Sbuf[64 * 64];     // 16KB; P aliases as bf16
    __shared__ float rowm[64], rowl[64], rowa[64];

    __bf16* Pb = (__bf16*)Sbuf;           // P row q at Pb[q*128 .. q*128+63]

    int tid  = threadIdx.x;
    int wave = tid >> 5;
    int lane = tid & 31;
    int half = lane >> 4;
    int l16  = lane & 15;

    auto issue_kv = [&](int kt, int kb) {
        for (int i = 0; i < 4; ++i) {
            int linear = (i * 128 + tid) * 8;
            async_b128(&Kp[(size_t)kt * 64 * HDD + linear], &Ks[kb][linear]);
            int d = linear >> 6, c = linear & 63;
            async_b128(&Vp[(size_t)d * SS + kt * 64 + c], &Vs[kb][linear]);
        }
    };

    // Q tile -> LDS (async), stats init
    for (int i = 0; i < 4; ++i) {
        int linear = (i * 128 + tid) * 8;
        async_b128(&Qp[linear], &Qs[linear]);
    }
    issue_kv(0, 0);
    if (tid < 64) { rowm[tid] = -3.0e38f; rowl[tid] = 0.f; }
    wait_async0();
    __syncthreads();

    Frag16 aq[2];
    {
        int row = wave * 16 + l16;
        for (int kk = 0; kk < 2; ++kk) {
            aq[kk].q[0] = *(uint4*)&Qs[row * 64 + kk * 32 + half * 8];
            aq[kk].q[1] = *(uint4*)&Qs[row * 64 + kk * 32 + 16 + half * 8];
        }
    }
    v8f ofrag[4] = {};
    int buf = 0;

    for (int kt = 0; kt <= qt; ++kt) {
        if (kt + 1 <= qt) issue_kv(kt + 1, buf ^ 1);   // DMA overlaps this tile

        // ---- scores: S = Q (16x64 strip) x K^T ----
        for (int nt = 0; nt < 4; ++nt) {
            v8f c = {};
            for (int kk = 0; kk < 2; ++kk) {
                Frag16 bk;
                int row = nt * 16 + l16;   // key row
                bk.q[0] = *(uint4*)&Ks[buf][row * 64 + kk * 32 + half * 8];
                bk.q[1] = *(uint4*)&Ks[buf][row * 64 + kk * 32 + 16 + half * 8];
                c = __builtin_amdgcn_wmma_f32_16x16x32_bf16(
                    false, aq[kk].v, false, bk.v, (short)0, c, false, false);
            }
            int kpos = kt * 64 + nt * 16 + l16;
            float pad = (amask[b * SS + kpos] > 0) ? 0.f : -1e10f;
            int mrow0 = wave * 16 + half * 8;
            for (int r = 0; r < 8; ++r) {
                int qpos = qt * 64 + mrow0 + r;
                float val = c[r] + pad;
                if (kpos > qpos) val += -1e10f;   // causal bias
                Sbuf[(mrow0 + r) * 64 + nt * 16 + l16] = val;
            }
        }
        __syncthreads();

        // ---- online softmax per row (threads 0..63, one row each) ----
        if (tid < 64) {
            float mold = rowm[tid];
            float mnew = mold;
            const float* srow = &Sbuf[tid * 64];
            for (int j = 0; j < 64; ++j) mnew = fmaxf(mnew, srow[j]);
            float alpha = __expf(mold - mnew);
            float sum = 0.f;
            __bf16* prow = &Pb[tid * 128];
            for (int j = 0; j < 64; ++j) {
                float pv = __expf(srow[j] - mnew);   // read before overwrite
                sum += pv;
                prow[j] = (__bf16)pv;                // destructive alias, safe
            }
            rowm[tid] = mnew;
            rowl[tid] = rowl[tid] * alpha + sum;
            rowa[tid] = alpha;
        }
        __syncthreads();

        // ---- O = O*alpha + P x V ----
        Frag16 ap[2];
        {
            int row = wave * 16 + l16;
            for (int kk = 0; kk < 2; ++kk) {
                ap[kk].q[0] = *(uint4*)&Pb[row * 128 + kk * 32 + half * 8];
                ap[kk].q[1] = *(uint4*)&Pb[row * 128 + kk * 32 + 16 + half * 8];
            }
        }
        int mrow0 = wave * 16 + half * 8;
        for (int nt = 0; nt < 4; ++nt) {
            v8f o = ofrag[nt];
            for (int r = 0; r < 8; ++r) o[r] *= rowa[mrow0 + r];
            for (int kk = 0; kk < 2; ++kk) {
                Frag16 bv;
                int row = nt * 16 + l16;   // hd row
                bv.q[0] = *(uint4*)&Vs[buf][row * 64 + kk * 32 + half * 8];
                bv.q[1] = *(uint4*)&Vs[buf][row * 64 + kk * 32 + 16 + half * 8];
                o = __builtin_amdgcn_wmma_f32_16x16x32_bf16(
                    false, ap[kk].v, false, bv.v, (short)0, o, false, false);
            }
            ofrag[nt] = o;
        }
        wait_async0();       // next K/V tile resident
        __syncthreads();     // and this tile's LDS reads are done everywhere
        buf ^= 1;
    }

    // ---- epilogue: normalize and scatter into [B,S,H*HD] bf16 ----
    int mrow0 = wave * 16 + half * 8;
    for (int nt = 0; nt < 4; ++nt) {
        for (int r = 0; r < 8; ++r) {
            int qpos = qt * 64 + mrow0 + r;
            float val = ofrag[nt][r] / rowl[mrow0 + r];
            obf[((size_t)b * SS + qpos) * DD + h * HDD + nt * 16 + l16] = (__bf16)val;
        }
    }
}

// =====================================================================
// host side
// =====================================================================
static inline void* arena(char* ws, size_t& off, size_t bytes) {
    off = (off + 255) & ~(size_t)255;
    void* p = ws + off;
    off += bytes;
    return p;
}

extern "C" void kernel_launch(void* const* d_in, const int* in_sizes, int n_in,
                              void* d_out, int out_size, void* d_ws, size_t ws_size,
                              hipStream_t stream) {
    (void)in_sizes; (void)n_in; (void)out_size; (void)ws_size;
    const float* x     = (const float*)d_in[0];
    const int*   amask = (const int*)d_in[1];
    const int*   pos   = (const int*)d_in[2];
    const float* ln1s  = (const float*)d_in[3];
    const float* ln1b  = (const float*)d_in[4];
    const float* wq    = (const float*)d_in[5];
    const float* wk    = (const float*)d_in[6];
    const float* wv    = (const float*)d_in[7];
    const float* wo    = (const float*)d_in[8];
    const float* ln2s  = (const float*)d_in[9];
    const float* ln2b  = (const float*)d_in[10];
    const float* w_in  = (const float*)d_in[11];
    const float* b_in  = (const float*)d_in[12];
    const float* w_out = (const float*)d_in[13];
    const float* b_out = (const float*)d_in[14];

    char* ws = (char*)d_ws;
    size_t off = 0;
    __bf16* wqT   = (__bf16*)arena(ws, off, (size_t)DD * DD * 2);
    __bf16* wkT   = (__bf16*)arena(ws, off, (size_t)DD * DD * 2);
    __bf16* wvT   = (__bf16*)arena(ws, off, (size_t)DD * DD * 2);
    __bf16* woT   = (__bf16*)arena(ws, off, (size_t)DD * DD * 2);
    __bf16* winT  = (__bf16*)arena(ws, off, (size_t)DD * FFD * 2);
    __bf16* woutT = (__bf16*)arena(ws, off, (size_t)DD * FFD * 2);
    __bf16* hbf   = (__bf16*)arena(ws, off, (size_t)MM * DD * 2);
    float*  qf    = (float*) arena(ws, off, (size_t)MM * DD * 4);
    float*  kf    = (float*) arena(ws, off, (size_t)MM * DD * 4);
    float*  vf    = (float*) arena(ws, off, (size_t)MM * DD * 4);
    __bf16* qbf   = (__bf16*)arena(ws, off, (size_t)MM * DD * 2);
    __bf16* kbf   = (__bf16*)arena(ws, off, (size_t)MM * DD * 2);
    __bf16* vtb   = (__bf16*)arena(ws, off, (size_t)MM * DD * 2);
    __bf16* obf   = (__bf16*)arena(ws, off, (size_t)MM * DD * 2);
    float*  x1    = (float*) arena(ws, off, (size_t)MM * DD * 4);
    __bf16* h2bf  = (__bf16*)arena(ws, off, (size_t)MM * DD * 2);
    __bf16* ffbf  = (__bf16*)arena(ws, off, (size_t)MM * FFD * 2);

    // weights -> bf16, transposed to [N][K]
    wcvt_kernel<<<dim3(DD / 32, DD / 32), 256, 0, stream>>>(wq, wqT, DD, DD);
    wcvt_kernel<<<dim3(DD / 32, DD / 32), 256, 0, stream>>>(wk, wkT, DD, DD);
    wcvt_kernel<<<dim3(DD / 32, DD / 32), 256, 0, stream>>>(wv, wvT, DD, DD);
    wcvt_kernel<<<dim3(DD / 32, DD / 32), 256, 0, stream>>>(wo, woT, DD, DD);
    wcvt_kernel<<<dim3(FFD / 32, DD / 32), 256, 0, stream>>>(w_in, winT, DD, FFD);
    wcvt_kernel<<<dim3(DD / 32, FFD / 32), 256, 0, stream>>>(w_out, woutT, FFD, DD);

    // LN1
    ln_kernel<<<MM, 256, 0, stream>>>(x, ln1s, ln1b, hbf, DD);

    // QKV projections
    gemm_kernel<false, false><<<dim3(DD / 128, MM / 128), 256, 0, stream>>>(
        hbf, wqT, nullptr, nullptr, qf, MM, DD, DD);
    gemm_kernel<false, false><<<dim3(DD / 128, MM / 128), 256, 0, stream>>>(
        hbf, wkT, nullptr, nullptr, kf, MM, DD, DD);
    gemm_kernel<false, false><<<dim3(DD / 128, MM / 128), 256, 0, stream>>>(
        hbf, wvT, nullptr, nullptr, vf, MM, DD, DD);

    // RoPE + reorder + bf16
    rope_kernel<<<MM, 256, 0, stream>>>(qf, kf, vf, pos, qbf, kbf, vtb);

    // flash attention
    attn_kernel<<<dim3(SS / 64, BB * HH), 128, 0, stream>>>(qbf, kbf, vtb, amask, obf);

    // O projection + residual  -> x1
    gemm_kernel<false, false><<<dim3(DD / 128, MM / 128), 256, 0, stream>>>(
        obf, woT, nullptr, x, x1, MM, DD, DD);

    // LN2
    ln_kernel<<<MM, 256, 0, stream>>>(x1, ln2s, ln2b, h2bf, DD);

    // FFN in (+bias, exact GELU) -> bf16
    gemm_kernel<true, true><<<dim3(FFD / 128, MM / 128), 256, 0, stream>>>(
        h2bf, winT, b_in, nullptr, ffbf, MM, FFD, DD);

    // FFN out (+bias, +residual x1) -> d_out fp32
    gemm_kernel<false, false><<<dim3(DD / 128, MM / 128), 256, 0, stream>>>(
        ffbf, woutT, b_out, x1, (float*)d_out, MM, DD, FFD);
}
